// TransformerConv_Model_40802189312831
// MI455X (gfx1250) — matrix operand order repaired and checked by reading, not compiled, not run on hardware
//
#include <hip/hip_runtime.h>

#define N_NODES  50000
#define N_EDGES  1600000
#define IN_CH    128
#define HEADS    4
#define OUT_CH   16
#define HC       64
#define N_GRAPHS 100
#define HIDDEN   128

typedef __attribute__((ext_vector_type(16))) _Float16 v16h;
typedef __attribute__((ext_vector_type(8)))  float    v8f;

// Sign-aware float atomic max via integer atomics (monotonic bit ordering).
__device__ __forceinline__ void atomicMaxF(float* addr, float val) {
    if (val >= 0.0f) atomicMax((int*)addr, __float_as_int(val));
    else             atomicMin((unsigned int*)addr, __float_as_uint(val));
}

// ---------------------------------------------------------------------------
// Pack Wcat = [Wq|Wk|Wv|Wskip] (f32 [128,64] each, row-major) into per-wave
// B fragments for v_wmma_f32_16x16x32_f16.
// B (32x16, 16-bit) per-lane layout: n = lane&15, kg = 16*(lane>>4),
// element e (0..15) holds B[kb + kg + e][n].
// Bpack index: ((kt*16 + nTile)*32 + lane)*16 f16 elements.
// Also builds the concatenated bias vector biascat[256].
// ---------------------------------------------------------------------------
__global__ void k_pack(const float* __restrict__ Wq, const float* __restrict__ Wk,
                       const float* __restrict__ Wv, const float* __restrict__ Ws,
                       const float* __restrict__ bq, const float* __restrict__ bk,
                       const float* __restrict__ bv, const float* __restrict__ bs,
                       _Float16* __restrict__ Bpack, float* __restrict__ biascat) {
    int tid = blockIdx.x * blockDim.x + threadIdx.x;
    if (tid < 256) {
        int w = tid >> 6, c = tid & 63;
        const float* b = (w == 0) ? bq : (w == 1) ? bk : (w == 2) ? bv : bs;
        biascat[tid] = b[c];
    }
    if (tid >= 4 * 16 * 32) return;
    int lane  = tid & 31;
    int nTile = (tid >> 5) & 15;
    int kt    = tid >> 9;                 // 0..3
    int n     = lane & 15;
    int kg    = (lane >> 4) * 16;
    int col   = nTile * 16 + n;           // 0..255 into Wcat
    int w     = col >> 6, wcol = col & 63;
    const float* W = (w == 0) ? Wq : (w == 1) ? Wk : (w == 2) ? Wv : Ws;
    _Float16* dst = Bpack + (size_t)tid * 16;
#pragma unroll
    for (int e = 0; e < 16; ++e) {
        int k = kt * 32 + kg + e;
        dst[e] = (_Float16)W[k * HC + wcol];
    }
}

// ---------------------------------------------------------------------------
// Init: agg/amax = -inf bits, den/pooled/cnt = 0.
// ---------------------------------------------------------------------------
__global__ void k_init(float* amax, float* den, float* agg, float* pooled, float* cnt) {
    int tid = blockIdx.x * blockDim.x + threadIdx.x;
    const unsigned NEG_INF = 0xff800000u;
    if (tid < N_NODES * HC)    ((unsigned*)agg)[tid] = NEG_INF;
    if (tid < N_NODES * HEADS) { ((unsigned*)amax)[tid] = NEG_INF; den[tid] = 0.0f; }
    if (tid < N_GRAPHS * HC)   pooled[tid] = 0.0f;
    if (tid < N_GRAPHS)        cnt[tid] = 0.0f;
}

// ---------------------------------------------------------------------------
// Fused projection GEMM: proj[50000,256] = x[50000,128] @ Wcat[128,256] + bias.
// One block = 16 waves; wave w computes the 16x16 tile at cols [16w,16w+16).
// A fragment (16-bit A 16x32): lane half=lane>>4, m=lane&15,
// elems 0..15 = K(kb+16*half .. +15), matching Bpack k order.
// ---------------------------------------------------------------------------
__global__ __launch_bounds__(512) void k_proj_gemm(const float* __restrict__ x,
                                                   const _Float16* __restrict__ Bpack,
                                                   const float* __restrict__ biascat,
                                                   float* __restrict__ proj) {
    const int lane = threadIdx.x & 31;
    const int wave = threadIdx.x >> 5;     // nTile 0..15
    const int half = lane >> 4;
    const int m    = lane & 15;
    const int m0   = blockIdx.x * 16;
    const float* arow = x + (size_t)(m0 + m) * IN_CH;
    v8f acc = {};
#pragma unroll
    for (int kt = 0; kt < 4; ++kt) {
        const int kb = kt * 32;
        float av[16];
        *(float4*)(av + 0)  = *(const float4*)(arow + kb + 16 * half);
        *(float4*)(av + 4)  = *(const float4*)(arow + kb + 16 * half + 4);
        *(float4*)(av + 8)  = *(const float4*)(arow + kb + 16 * half + 8);
        *(float4*)(av + 12) = *(const float4*)(arow + kb + 16 * half + 12);
        v16h a;
#pragma unroll
        for (int i = 0; i < 16; ++i) a[i] = (_Float16)av[i];
        v16h b = *(const v16h*)(Bpack + ((((size_t)kt * 16 + wave) * 32) + lane) * 16);
        acc = __builtin_amdgcn_wmma_f32_16x16x32_f16(
            /*neg_a=*/false, a, /*neg_b=*/false, b,
            /*c_mod=*/(short)0, acc, /*reuse_a=*/false, /*reuse_b=*/false);
    }
    // D layout: VGPR r -> row m0 + 8*half + r, col = 16*wave + (lane&15)
    const int col  = wave * 16 + (lane & 15);
    const float bc = biascat[col];
    const int rowb = m0 + 8 * half;
#pragma unroll
    for (int r = 0; r < 8; ++r)
        proj[(size_t)(rowb + r) * 256 + col] = acc[r] + bc;
}

// ---------------------------------------------------------------------------
// Pass 1: alpha[e,h] = <q[dst], k[src]+ea*We> / sqrt(16); segment max into amax.
// proj layout: [q(0:64) | k(64:128) | v(128:192) | skip(192:256)]
// ---------------------------------------------------------------------------
__global__ void k_edge_alpha(const int* __restrict__ ei, const float* __restrict__ ea,
                             const float* __restrict__ We, const float* __restrict__ proj,
                             float* __restrict__ alphaB, float* __restrict__ amax) {
    int tid = blockIdx.x * blockDim.x + threadIdx.x;
    if (tid >= N_EDGES * HEADS) return;
    int e = tid >> 2, h = tid & 3;
    int src = ei[e], dst = ei[N_EDGES + e];
    float w = ea[e];
    const float4* qp = (const float4*)(proj + (size_t)dst * 256 + h * 16);
    const float4* kp = (const float4*)(proj + (size_t)src * 256 + 64 + h * 16);
    const float4* wp = (const float4*)(We + h * 16);
    float dot = 0.0f;
#pragma unroll
    for (int i = 0; i < 4; ++i) {
        float4 q = qp[i], k = kp[i], s = wp[i];
        dot += q.x * (k.x + w * s.x) + q.y * (k.y + w * s.y)
             + q.z * (k.z + w * s.z) + q.w * (k.w + w * s.w);
    }
    float alpha = dot * 0.25f;               // 1/sqrt(OUT_CH)
    alphaB[tid] = alpha;
    atomicMaxF(&amax[dst * HEADS + h], alpha);
}

// ---------------------------------------------------------------------------
// Pass 2: ex = exp(alpha - amax[dst]); segment sum into den.
// ---------------------------------------------------------------------------
__global__ void k_edge_exp(const int* __restrict__ ei, float* __restrict__ alphaB,
                           const float* __restrict__ amax, float* __restrict__ den) {
    int tid = blockIdx.x * blockDim.x + threadIdx.x;
    if (tid >= N_EDGES * HEADS) return;
    int e = tid >> 2, h = tid & 3;
    int dst = ei[N_EDGES + e];
    float ex = __expf(alphaB[tid] - amax[dst * HEADS + h]);
    alphaB[tid] = ex;
    atomicAdd(&den[dst * HEADS + h], ex);
}

// ---------------------------------------------------------------------------
// Pass 3: msg = (v[src]+ea*We) * ex/den[dst]; scatter-MAX into agg[dst].
// ---------------------------------------------------------------------------
__global__ void k_edge_msg(const int* __restrict__ ei, const float* __restrict__ ea,
                           const float* __restrict__ We, const float* __restrict__ proj,
                           const float* __restrict__ alphaB, const float* __restrict__ den,
                           float* __restrict__ agg) {
    int tid = blockIdx.x * blockDim.x + threadIdx.x;
    if (tid >= N_EDGES * HEADS) return;
    int e = tid >> 2, h = tid & 3;
    int src = ei[e], dst = ei[N_EDGES + e];
    float a = alphaB[tid] / fmaxf(den[dst * HEADS + h], 1e-16f);
    float w = ea[e];
    const float4* vp = (const float4*)(proj + (size_t)src * 256 + 128 + h * 16);
    const float4* wp = (const float4*)(We + h * 16);
    float* ag = agg + (size_t)dst * HC + h * 16;
#pragma unroll
    for (int i = 0; i < 4; ++i) {
        float4 v = vp[i], s = wp[i];
        atomicMaxF(&ag[4 * i + 0], (v.x + w * s.x) * a);
        atomicMaxF(&ag[4 * i + 1], (v.y + w * s.y) * a);
        atomicMaxF(&ag[4 * i + 2], (v.z + w * s.z) * a);
        atomicMaxF(&ag[4 * i + 3], (v.w + w * s.w) * a);
    }
}

// ---------------------------------------------------------------------------
// Node finalize: out = where(finite, agg, 0) + skip; mean-pool accumulate.
// ---------------------------------------------------------------------------
__global__ void k_node_final(const float* __restrict__ agg, const float* __restrict__ proj,
                             const int* __restrict__ batch,
                             float* __restrict__ pooled, float* __restrict__ cnt) {
    int tid = blockIdx.x * blockDim.x + threadIdx.x;
    if (tid >= N_NODES * HC) return;
    int n = tid >> 6, c = tid & 63;
    float a = agg[tid];
    if (!(fabsf(a) < __builtin_inff())) a = 0.0f;   // non-finite -> 0 (NaN/±inf)
    float o = a + proj[(size_t)n * 256 + 192 + c];  // skip already has bias
    int g = batch[n];
    atomicAdd(&pooled[g * HC + c], o);
    if (c == 0) atomicAdd(&cnt[g], 1.0f);
}

// ---------------------------------------------------------------------------
// Per-graph MLP head: relu(pooled/cnt @ W1 + b1) @ W2 + b2.
// ---------------------------------------------------------------------------
__global__ __launch_bounds__(128) void k_mlp(const float* __restrict__ pooled,
                                             const float* __restrict__ cnt,
                                             const float* __restrict__ W1, const float* __restrict__ b1,
                                             const float* __restrict__ W2, const float* __restrict__ b2,
                                             float* __restrict__ out) {
    __shared__ float sp[HC];
    __shared__ float sh[HIDDEN];
    int g = blockIdx.x, t = threadIdx.x;
    float inv = 1.0f / fmaxf(cnt[g], 1.0f);
    if (t < HC) sp[t] = pooled[g * HC + t] * inv;
    __syncthreads();
    float acc = b1[t];
#pragma unroll 8
    for (int c = 0; c < HC; ++c) acc += sp[c] * W1[c * HIDDEN + t];
    sh[t] = fmaxf(acc, 0.0f);
    __syncthreads();
    if (t < 2) {
        float o = b2[t];
        for (int j = 0; j < HIDDEN; ++j) o += sh[j] * W2[j * 2 + t];
        out[g * 2 + t] = o;
    }
}

extern "C" void kernel_launch(void* const* d_in, const int* in_sizes, int n_in,
                              void* d_out, int out_size, void* d_ws, size_t ws_size,
                              hipStream_t stream) {
    const float* x   = (const float*)d_in[0];
    const int*   ei  = (const int*)d_in[1];
    const float* ea  = (const float*)d_in[2];
    const int*   bat = (const int*)d_in[3];
    const float* Wq  = (const float*)d_in[4];
    const float* bq  = (const float*)d_in[5];
    const float* Wk  = (const float*)d_in[6];
    const float* bk  = (const float*)d_in[7];
    const float* Wv  = (const float*)d_in[8];
    const float* bv  = (const float*)d_in[9];
    const float* We  = (const float*)d_in[10];
    const float* Wsk = (const float*)d_in[11];
    const float* bsk = (const float*)d_in[12];
    const float* W1  = (const float*)d_in[13];
    const float* b1  = (const float*)d_in[14];
    const float* W2  = (const float*)d_in[15];
    const float* b2  = (const float*)d_in[16];
    float* out = (float*)d_out;

    char* ws = (char*)d_ws;
    size_t off = 0;
    auto carve = [&](size_t bytes) -> char* {
        char* p = ws + off;
        off = (off + bytes + 255) & ~(size_t)255;
        return p;
    };
    _Float16* Bpack  = (_Float16*)carve((size_t)4 * 16 * 32 * 16 * sizeof(_Float16)); // 64 KB
    float* biascat   = (float*)carve(256 * sizeof(float));
    float* proj      = (float*)carve((size_t)N_NODES * 256 * sizeof(float));          // 51.2 MB
    float* alphaB    = (float*)carve((size_t)N_EDGES * HEADS * sizeof(float));        // 25.6 MB
    float* amax      = (float*)carve((size_t)N_NODES * HEADS * sizeof(float));
    float* den       = (float*)carve((size_t)N_NODES * HEADS * sizeof(float));
    float* agg       = (float*)carve((size_t)N_NODES * HC * sizeof(float));           // 12.8 MB
    float* pooled    = (float*)carve((size_t)N_GRAPHS * HC * sizeof(float));
    float* cnt       = (float*)carve((size_t)N_GRAPHS * sizeof(float));
    (void)ws_size; (void)in_sizes; (void)n_in; (void)out_size;

    const int EH = N_EDGES * HEADS;

    k_pack<<<8, 256, 0, stream>>>(Wq, Wk, Wv, Wsk, bq, bk, bv, bsk, Bpack, biascat);
    k_init<<<(N_NODES * HC + 255) / 256, 256, 0, stream>>>(amax, den, agg, pooled, cnt);
    k_proj_gemm<<<N_NODES / 16, 512, 0, stream>>>(x, Bpack, biascat, proj);
    k_edge_alpha<<<(EH + 255) / 256, 256, 0, stream>>>(ei, ea, We, proj, alphaB, amax);
    k_edge_exp<<<(EH + 255) / 256, 256, 0, stream>>>(ei, alphaB, amax, den);
    k_edge_msg<<<(EH + 255) / 256, 256, 0, stream>>>(ei, ea, We, proj, alphaB, den, agg);
    k_node_final<<<(N_NODES * HC + 255) / 256, 256, 0, stream>>>(agg, proj, bat, pooled, cnt);
    k_mlp<<<N_GRAPHS, 128, 0, stream>>>(pooled, cnt, W1, b1, W2, b2, out);
}
